// LoRALinear_53008486367361
// MI455X (gfx1250) — matrix operand order repaired and checked by reading, not compiled, not run on hardware
//
#include <hip/hip_runtime.h>
#include <hip/hip_bf16.h>

typedef __attribute__((ext_vector_type(16))) _Float16 v16h;
typedef __attribute__((ext_vector_type(8)))  _Float16 v8h;
typedef __attribute__((ext_vector_type(8)))  float    v8f;
typedef __attribute__((ext_vector_type(4)))  unsigned int v4u;
typedef __attribute__((ext_vector_type(8)))  int      v8i;
typedef __attribute__((ext_vector_type(4)))  int      v4i;

#define N_ROWS  16384
#define D_IN    1024
#define D_OUT   1024
#define RANK    8
#define SCALING 4.0f

// GEMM tiling
#define BM 128      // block tile M (rows of x)
#define BN 128      // block tile N (cols of out / rows of W')
#define KC 32       // K chunk per LDS stage (one WMMA K-step)
#define LDS_S 40    // halfs per LDS row: 32 data + 8 pad (80B, 16B aligned)

#if defined(__gfx1250__) && __has_builtin(__builtin_amdgcn_tensor_load_to_lds) && \
    __has_builtin(__builtin_amdgcn_s_wait_tensorcnt)
#define USE_TDM 1
#else
#define USE_TDM 0
#endif

// ---------------------------------------------------------------------------
// Kernel 1: convert x (f32) -> f16, 4 elements per thread
// ---------------------------------------------------------------------------
__global__ void lora_cvt_x(const float* __restrict__ x, _Float16* __restrict__ xh) {
    int i = (blockIdx.x * blockDim.x + threadIdx.x) * 4;
    float4 v = *(const float4*)(x + i);
    _Float16* p = xh + i;
    p[0] = (_Float16)v.x; p[1] = (_Float16)v.y;
    p[2] = (_Float16)v.z; p[3] = (_Float16)v.w;
}

// ---------------------------------------------------------------------------
// Kernel 2: W' = W + SCALING * (B @ A), stored f16  (D_OUT x D_IN, row major)
// ---------------------------------------------------------------------------
__global__ void lora_fuse_w(const float* __restrict__ W,
                            const float* __restrict__ A,
                            const float* __restrict__ B,
                            _Float16* __restrict__ wh) {
    int idx = blockIdx.x * blockDim.x + threadIdx.x;   // 0 .. 1M-1
    int o = idx >> 10;
    int d = idx & (D_IN - 1);
    float s = 0.0f;
#pragma unroll
    for (int r = 0; r < RANK; ++r)
        s += B[o * RANK + r] * A[r * D_IN + d];
    wh[idx] = (_Float16)(W[idx] + SCALING * s);
}

#if USE_TDM
// ---------------------------------------------------------------------------
// TDM: load a (16 rows x 32 halfs) sub-tile from a row-major f16 matrix into
// LDS at lds_byte_addr, injecting 16B padding after every 64B row so the LDS
// image has an 80B (LDS_S) row stride.
//   D# group0: count=1 | lds_addr | global_addr(57b) | type=2
//   D# group1: data_size=2B, pad_enable, pad_interval=3 (16 DW), pad_amount=3
//              (4 DW), tensor_dim0=D_IN, tile_dim0=KC, tile_dim1=16,
//              tensor_dim0_stride=D_IN
// ---------------------------------------------------------------------------
__device__ __forceinline__
void tdm_load_tile16(const _Float16* gbase, unsigned long long row0, int kt,
                     unsigned int rows_total, unsigned int lds_byte_addr) {
    unsigned long long ga = (unsigned long long)(uintptr_t)gbase +
                            (row0 * (unsigned long long)D_IN + (unsigned)kt) * 2ull;
    v4u g0;
    g0.x = 1u;                                          // count=1, user mode
    g0.y = lds_byte_addr;                               // lds_addr [63:32]
    g0.z = (unsigned int)(ga & 0xFFFFFFFFull);          // global_addr lo
    g0.w = (unsigned int)((ga >> 32) & 0x01FFFFFFull)   // global_addr hi (bits 120:96)
         | (2u << 30);                                  // type=2 ("image")
    v8i g1;
    g1[0] = (1 << 16)       // data_size = 2 bytes
          | (1 << 20)       // pad_enable
          | (3 << 22)       // pad_interval: 16 DWORDs (64B) of data
          | (3 << 25);      // pad_amount:   4 DWORDs (16B) of pad
    g1[1] = (int)((unsigned)D_IN << 16);          // tensor_dim0[15:0] in [31:16]
    g1[2] = (int)((rows_total & 0xFFFFu) << 16);  // tensor_dim0 hi=0 | tensor_dim1 lo
    g1[3] = (KC << 16);                           // tensor_dim1 hi=0 | tile_dim0=32
    g1[4] = 16;                                   // tile_dim1=16 rows, tile_dim2=0
    g1[5] = D_IN;                                 // tensor_dim0_stride lo32
    g1[6] = 0;                                    // stride hi | dim1_stride lo
    g1[7] = 0;
    v4i z4 = {0, 0, 0, 0};
#if __clang_major__ >= 23
    v8i z8 = {0, 0, 0, 0, 0, 0, 0, 0};
    __builtin_amdgcn_tensor_load_to_lds(g0, g1, z4, z4, z8, 0);
#else
    __builtin_amdgcn_tensor_load_to_lds(g0, g1, z4, z4, 0);
#endif
}
#endif

// ---------------------------------------------------------------------------
// Kernel 3: out[n,o] = sum_d xh[n,d] * wh[o,d] + bias[o]
// 256 threads = 8 waves (wave32), wave grid 2(M) x 4(N); per-wave 64x32 tile.
// TDM path: double-buffered LDS, each wave TDM-loads its own 16-row sub-tile
// of the next K-slab while computing WMMAs on the current one.
// ---------------------------------------------------------------------------
__global__ __launch_bounds__(256)
void lora_gemm_f16(const _Float16* __restrict__ xh,
                   const _Float16* __restrict__ wh,
                   const float* __restrict__ bias,
                   float* __restrict__ out) {
#if USE_TDM
    __shared__ _Float16 sx[2][BM * LDS_S];
    __shared__ _Float16 sw[2][BN * LDS_S];
#else
    __shared__ _Float16 sx[1][BM * LDS_S];
    __shared__ _Float16 sw[1][BN * LDS_S];
#endif

    const int tid  = threadIdx.x;
    const int lane = tid & 31;
#if USE_TDM
    const int wave = __builtin_amdgcn_readfirstlane(tid >> 5);
#else
    const int wave = tid >> 5;
#endif
    const int wm   = wave & 1;   // 0..1  -> 64-row M strip
    const int wn   = wave >> 1;  // 0..3  -> 32-col N strip

    const int bm = blockIdx.y;   // 0..127
    const int bn = blockIdx.x;   // 0..7

    const int lr = lane & 15;
    const int hi = lane >> 4;

    v8f acc[4][2];
#pragma unroll
    for (int mi = 0; mi < 4; ++mi)
#pragma unroll
        for (int ni = 0; ni < 2; ++ni)
            acc[mi][ni] = (v8f)(0.0f);

#if USE_TDM
    // ---- preload K-slab 0 via Tensor Data Mover ---------------------------
    {
        unsigned int lx = (unsigned int)(uintptr_t)&sx[0][wave * 16 * LDS_S];
        unsigned int lw = (unsigned int)(uintptr_t)&sw[0][wave * 16 * LDS_S];
        tdm_load_tile16(xh, (unsigned long long)bm * BM + wave * 16, 0, N_ROWS, lx);
        tdm_load_tile16(wh, (unsigned long long)bn * BN + wave * 16, 0, D_OUT, lw);
    }
    __builtin_amdgcn_s_wait_tensorcnt((short)0);
    __syncthreads();
#endif

    for (int kt = 0; kt < D_IN; kt += KC) {
#if USE_TDM
        const int cur = (kt >> 5) & 1;
        // ---- async-stage next K-slab into the other buffer ----------------
        if (kt + KC < D_IN) {
            unsigned int lx = (unsigned int)(uintptr_t)&sx[cur ^ 1][wave * 16 * LDS_S];
            unsigned int lw = (unsigned int)(uintptr_t)&sw[cur ^ 1][wave * 16 * LDS_S];
            tdm_load_tile16(xh, (unsigned long long)bm * BM + wave * 16, kt + KC, N_ROWS, lx);
            tdm_load_tile16(wh, (unsigned long long)bn * BN + wave * 16, kt + KC, D_OUT, lw);
        }
#else
        const int cur = 0;
        // ---- synchronous staging fallback ---------------------------------
#pragma unroll
        for (int i = 0; i < 2; ++i) {
            int c   = tid + 256 * i;
            int row = c >> 2;
            int off = (c & 3) * 8;
            const v8h* gx = (const v8h*)(xh + (size_t)(bm * BM + row) * D_IN + kt + off);
            *(v8h*)(&sx[0][row * LDS_S + off]) = *gx;
            const v8h* gw = (const v8h*)(wh + (size_t)(bn * BN + row) * D_IN + kt + off);
            *(v8h*)(&sw[0][row * LDS_S + off]) = *gw;
        }
        __syncthreads();
        if (kt + KC < D_IN) {
            int row = tid >> 2;
            int off = (tid & 3) * 8;
            __builtin_prefetch(xh + (size_t)(bm * BM + row) * D_IN + kt + KC + off, 0, 0);
            __builtin_prefetch(wh + (size_t)(bn * BN + row) * D_IN + kt + KC + off, 0, 0);
        }
#endif

        // ---- load fragments per CDNA5 WMMA VGPR layouts -------------------
        // A (16x32 f16): lane holds M=lane%16; K chunks [hi*8,+8) and [16+hi*8,+8)
        v16h af[4];
#pragma unroll
        for (int mi = 0; mi < 4; ++mi) {
            int row = wm * 64 + mi * 16 + lr;
            v8h lo = *(const v8h*)(&sx[cur][row * LDS_S + hi * 8]);
            v8h hh = *(const v8h*)(&sx[cur][row * LDS_S + 16 + hi * 8]);
            af[mi] = __builtin_shufflevector(lo, hh,
                0,1,2,3,4,5,6,7,8,9,10,11,12,13,14,15);
        }
        // B (32x16 f16): lane holds N=lane%16; K chunks [hi*8,+8) and [16+hi*8,+8)
        v16h bf[2];
#pragma unroll
        for (int ni = 0; ni < 2; ++ni) {
            int row = wn * 32 + ni * 16 + lr;
            v8h lo = *(const v8h*)(&sw[cur][row * LDS_S + hi * 8]);
            v8h hh = *(const v8h*)(&sw[cur][row * LDS_S + 16 + hi * 8]);
            bf[ni] = __builtin_shufflevector(lo, hh,
                0,1,2,3,4,5,6,7,8,9,10,11,12,13,14,15);
        }

        // ---- 8x v_wmma_f32_16x16x32_f16 -----------------------------------
#pragma unroll
        for (int mi = 0; mi < 4; ++mi)
#pragma unroll
            for (int ni = 0; ni < 2; ++ni)
                acc[mi][ni] = __builtin_amdgcn_wmma_f32_16x16x32_f16(
                    false, af[mi], false, bf[ni],
                    (short)0, acc[mi][ni], false, false);

#if USE_TDM
        __builtin_amdgcn_s_wait_tensorcnt((short)0);  // next slab landed
#endif
        __syncthreads();
    }

    // ---- epilogue: add bias, store f32 ------------------------------------
#pragma unroll
    for (int ni = 0; ni < 2; ++ni) {
        int n = bn * BN + wn * 32 + ni * 16 + lr;
        float bv = bias[n];
#pragma unroll
        for (int mi = 0; mi < 4; ++mi) {
#pragma unroll
            for (int v = 0; v < 8; ++v) {
                int m = bm * BM + wm * 64 + mi * 16 + v + 8 * hi;
                out[(size_t)m * D_OUT + n] = acc[mi][ni][v] + bv;
            }
        }
    }
}

// ---------------------------------------------------------------------------
// Host-side launcher
// ---------------------------------------------------------------------------
extern "C" void kernel_launch(void* const* d_in, const int* in_sizes, int n_in,
                              void* d_out, int out_size, void* d_ws, size_t ws_size,
                              hipStream_t stream) {
    const float* x    = (const float*)d_in[0];   // [N, D_IN]
    const float* W    = (const float*)d_in[1];   // [D_OUT, D_IN]
    const float* bias = (const float*)d_in[2];   // [D_OUT]
    const float* A    = (const float*)d_in[3];   // [R, D_IN]
    const float* B    = (const float*)d_in[4];   // [D_OUT, R]
    float* out = (float*)d_out;

    _Float16* xh = (_Float16*)d_ws;                                      // 32 MB
    _Float16* wh = (_Float16*)((char*)d_ws + (size_t)N_ROWS * D_IN * 2); // +2 MB

    {
        int total = N_ROWS * D_IN;
        lora_cvt_x<<<total / (256 * 4), 256, 0, stream>>>(x, xh);
    }
    {
        int total = D_OUT * D_IN;
        lora_fuse_w<<<total / 256, 256, 0, stream>>>(W, A, B, wh);
    }
    {
        dim3 grid(D_OUT / BN, N_ROWS / BM);  // (8, 128)
        lora_gemm_f16<<<grid, 256, 0, stream>>>(xh, wh, bias, out);
    }
}
